// RBM_17806934409607
// MI455X (gfx1250) — compile-verified, matrix operand order checked by reading, not run on hardware
//
#include <hip/hip_runtime.h>

// ---------------------------------------------------------------------------
// RBM CD-1 step for MI455X (gfx1250, wave32, WMMA).
// All six 8192x2048x1024-class GEMMs run on v_wmma_f32_16x16x32_f16 with f32
// accumulation. Compute-bound workload (~206 GFLOP vs ~250MB traffic), so we
// spend a little extra bandwidth making f16 row-major [rows,K] copies of every
// operand (incl. transposes) so each wave's fragment loads are two contiguous
// b128 loads per lane, matching the ISA 16-bit A/B VGPR layout.
// ---------------------------------------------------------------------------

#define DEVINL __device__ __forceinline__

typedef __attribute__((ext_vector_type(16))) _Float16 f16x16;
typedef __attribute__((ext_vector_type(8)))  _Float16 f16x8;
typedef __attribute__((ext_vector_type(8)))  float    f32x8;

static constexpr int Bn = 8192;   // batch
static constexpr int Vn = 2048;   // visible
static constexpr int Hn = 1024;   // hidden

// d_out layout (floats): h_pred [B*H] | recon [1] | w_grad [V*H] | vb [V] | hb [H]
static constexpr size_t OFF_HP    = 0;
static constexpr size_t OFF_RECON = (size_t)Bn * Hn;
static constexpr size_t OFF_WG    = OFF_RECON + 1;
static constexpr size_t OFF_VB    = OFF_WG + (size_t)Vn * Hn;
static constexpr size_t OFF_HB    = OFF_VB + Vn;

// --- fragment load: operand stored [rows, K] row-major f16 ------------------
// A-matrix 16x32 f16 layout (ISA 7.12.2): lanes 0-15 hold row M=lane,
// K in {0..7, 16..23}; lanes 16-31 hold row M=lane-16, K in {8..15, 24..31}.
// B fragments use the identical pattern with "row" = output column (since we
// store B pre-transposed as [N, K]).
DEVINL f16x16 load_frag(const _Float16* __restrict__ base, int row, int ld,
                        int kb, int lane) {
  const int half = lane >> 4;
  const _Float16* p = base + (size_t)row * ld + kb + half * 8;
  f16x8 lo = *(const f16x8*)(p);
  f16x8 hi = *(const f16x8*)(p + 16);
  return __builtin_shufflevector(lo, hi, 0,1,2,3,4,5,6,7,8,9,10,11,12,13,14,15);
}

DEVINL f32x8 wmma16(f16x16 a, f16x16 b, f32x8 c) {
  // (neg_a, A, neg_b, B, c_mod, C, reuse_a, reuse_b)
  return __builtin_amdgcn_wmma_f32_16x16x32_f16(false, a, false, b, (short)0, c,
                                                false, false);
}

DEVINL float sigmoidf(float x) { return 1.0f / (1.0f + __expf(-x)); }

// --- elementwise f32 -> f16 -------------------------------------------------
__global__ void k_cvt_f16(const float* __restrict__ in, _Float16* __restrict__ out,
                          size_t n) {
  size_t i = (size_t)blockIdx.x * blockDim.x + threadIdx.x;
  size_t stride = (size_t)gridDim.x * blockDim.x;
  for (; i < n; i += stride) out[i] = (_Float16)in[i];
}

// --- LDS-tiled transpose (+convert) to f16: in[R,C] -> out[C,R] -------------
template <typename Tin>
__global__ void k_transpose_f16(const Tin* __restrict__ in,
                                _Float16* __restrict__ out, int R, int C) {
  __shared__ _Float16 tile[32][33];  // +1 pad vs LDS bank conflicts
  const int c0 = blockIdx.x * 32;
  const int r0 = blockIdx.y * 32;
  for (int j = threadIdx.y; j < 32; j += 8)
    tile[j][threadIdx.x] =
        (_Float16)(float)in[(size_t)(r0 + j) * C + (c0 + threadIdx.x)];
  __syncthreads();
  for (int j = threadIdx.y; j < 32; j += 8)
    out[(size_t)(c0 + j) * R + (r0 + threadIdx.x)] = tile[threadIdx.x][j];
}

// --- GEMM 1/3: C = sigmoid(hb + A@w) [+ sample] ----------------------------
// A: [B, K] f16 ; Bt: [H, K] f16 (w transposed to [N,K]) ; M=Bn, N=Hn.
// MODE 0 (mean-field forward): writes h_pred f32, hp f16, h0 = (p > u).
// MODE 1 (chain step):         writes h1 = (p > u) only.
template <int MODE>
__global__ __launch_bounds__(256)
void k_gemm_h(const _Float16* __restrict__ A, const _Float16* __restrict__ Bt,
              const float* __restrict__ hb, const float* __restrict__ u,
              float* __restrict__ hp_out, _Float16* __restrict__ hp_f16,
              _Float16* __restrict__ hbin_f16, int K) {
  const int lane = threadIdx.x & 31;
  const int wid  = threadIdx.x >> 5;                 // 8 waves: 4(M) x 2(N)
  const int m0 = blockIdx.y * 128 + (wid & 3) * 32;
  const int n0 = blockIdx.x * 64  + (wid >> 2) * 32;
  const int r  = lane & 15;

  f32x8 acc[2][2] = {};
  for (int kb = 0; kb < K; kb += 32) {
    if (kb + 32 < K) {  // emits global_prefetch_b8
      __builtin_prefetch(A  + (size_t)(m0 + r) * K + kb + 32, 0, 3);
      __builtin_prefetch(Bt + (size_t)(n0 + r) * K + kb + 32, 0, 3);
    }
    f16x16 a0 = load_frag(A,  m0 + r,      K, kb, lane);
    f16x16 a1 = load_frag(A,  m0 + 16 + r, K, kb, lane);
    f16x16 b0 = load_frag(Bt, n0 + r,      K, kb, lane);
    f16x16 b1 = load_frag(Bt, n0 + 16 + r, K, kb, lane);
    acc[0][0] = wmma16(a0, b0, acc[0][0]);
    acc[0][1] = wmma16(a0, b1, acc[0][1]);
    acc[1][0] = wmma16(a1, b0, acc[1][0]);
    acc[1][1] = wmma16(a1, b1, acc[1][1]);
  }

  // C layout: VGPR e, lanes 0-15 -> row e, lanes 16-31 -> row e+8; col = lane&15
  const int half = lane >> 4;
  for (int mi = 0; mi < 2; ++mi)
    for (int ni = 0; ni < 2; ++ni)
      for (int e = 0; e < 8; ++e) {
        const int row = m0 + mi * 16 + half * 8 + e;
        const int col = n0 + ni * 16 + r;
        const size_t idx = (size_t)row * Hn + col;
        const float p = sigmoidf(acc[mi][ni][e] + hb[col]);
        if (MODE == 0) {
          hp_out[idx] = p;
          hp_f16[idx] = (_Float16)p;
        }
        hbin_f16[idx] = (p > u[idx]) ? (_Float16)1.0f : (_Float16)0.0f;
      }
}

// --- GEMM 2 (dual): v_pred = sig(vb + hp@wT), vk = sample(sig(vb + h0@wT)) --
// Shares the B (w) fragment loads between the two accumulator sets.
__global__ __launch_bounds__(256)
void k_gemm_vk(const _Float16* __restrict__ Ahp, const _Float16* __restrict__ Ah0,
               const _Float16* __restrict__ Bw,   // w f16 [V,H] == [N,K]
               const float* __restrict__ vb, const float* __restrict__ v,
               const float* __restrict__ u_v0, _Float16* __restrict__ vk_f16,
               float* __restrict__ recon, int K) {
  const int lane = threadIdx.x & 31;
  const int wid  = threadIdx.x >> 5;
  const int m0 = blockIdx.y * 128 + (wid & 3) * 32;
  const int n0 = blockIdx.x * 64  + (wid >> 2) * 32;
  const int r  = lane & 15;

  f32x8 acc1[2][2] = {};  // hp path (v_pred)
  f32x8 acc2[2][2] = {};  // h0 path (vk)
  for (int kb = 0; kb < K; kb += 32) {
    f16x16 b0 = load_frag(Bw, n0 + r,      K, kb, lane);
    f16x16 b1 = load_frag(Bw, n0 + 16 + r, K, kb, lane);
    f16x16 p0 = load_frag(Ahp, m0 + r,      K, kb, lane);
    f16x16 p1 = load_frag(Ahp, m0 + 16 + r, K, kb, lane);
    f16x16 s0 = load_frag(Ah0, m0 + r,      K, kb, lane);
    f16x16 s1 = load_frag(Ah0, m0 + 16 + r, K, kb, lane);
    acc1[0][0] = wmma16(p0, b0, acc1[0][0]);
    acc1[0][1] = wmma16(p0, b1, acc1[0][1]);
    acc1[1][0] = wmma16(p1, b0, acc1[1][0]);
    acc1[1][1] = wmma16(p1, b1, acc1[1][1]);
    acc2[0][0] = wmma16(s0, b0, acc2[0][0]);
    acc2[0][1] = wmma16(s0, b1, acc2[0][1]);
    acc2[1][0] = wmma16(s1, b0, acc2[1][0]);
    acc2[1][1] = wmma16(s1, b1, acc2[1][1]);
  }

  const int half = lane >> 4;
  float err = 0.0f;
  for (int mi = 0; mi < 2; ++mi)
    for (int ni = 0; ni < 2; ++ni)
      for (int e = 0; e < 8; ++e) {
        const int row = m0 + mi * 16 + half * 8 + e;
        const int col = n0 + ni * 16 + r;
        const size_t idx = (size_t)row * Vn + col;
        const float bias = vb[col];
        const float vp = sigmoidf(acc1[mi][ni][e] + bias);
        err += fabsf(v[idx] - vp);
        const float pk = sigmoidf(acc2[mi][ni][e] + bias);
        vk_f16[idx] = (pk > u_v0[idx]) ? (_Float16)1.0f : (_Float16)0.0f;
      }

  __shared__ float red[256];
  red[threadIdx.x] = err;
  __syncthreads();
  for (int s = 128; s > 0; s >>= 1) {
    if ((int)threadIdx.x < s) red[threadIdx.x] += red[threadIdx.x + s];
    __syncthreads();
  }
  if (threadIdx.x == 0)
    atomicAdd(recon, red[0] * (1.0f / ((float)Bn * (float)Vn)));
}

// --- GEMM 4 (dual): w_grad = vk^T @ h1 - v^T @ h0 ---------------------------
// Operands are the transposed f16 copies: A* [V,B], B* [H,B]; M=Vn, N=Hn, K=Bn.
__global__ __launch_bounds__(256)
void k_gemm_wgrad(const _Float16* __restrict__ Avk, const _Float16* __restrict__ Av,
                  const _Float16* __restrict__ Bh1, const _Float16* __restrict__ Bh0,
                  float* __restrict__ wgrad, int K) {
  const int lane = threadIdx.x & 31;
  const int wid  = threadIdx.x >> 5;
  const int m0 = blockIdx.y * 128 + (wid & 3) * 32;
  const int n0 = blockIdx.x * 64  + (wid >> 2) * 32;
  const int r  = lane & 15;

  f32x8 accM[2][2] = {};  // model term
  f32x8 accD[2][2] = {};  // data term
  for (int kb = 0; kb < K; kb += 32) {
    f16x16 ak0 = load_frag(Avk, m0 + r,      K, kb, lane);
    f16x16 ak1 = load_frag(Avk, m0 + 16 + r, K, kb, lane);
    f16x16 av0 = load_frag(Av,  m0 + r,      K, kb, lane);
    f16x16 av1 = load_frag(Av,  m0 + 16 + r, K, kb, lane);
    f16x16 b10 = load_frag(Bh1, n0 + r,      K, kb, lane);
    f16x16 b11 = load_frag(Bh1, n0 + 16 + r, K, kb, lane);
    f16x16 b00 = load_frag(Bh0, n0 + r,      K, kb, lane);
    f16x16 b01 = load_frag(Bh0, n0 + 16 + r, K, kb, lane);
    accM[0][0] = wmma16(ak0, b10, accM[0][0]);
    accM[0][1] = wmma16(ak0, b11, accM[0][1]);
    accM[1][0] = wmma16(ak1, b10, accM[1][0]);
    accM[1][1] = wmma16(ak1, b11, accM[1][1]);
    accD[0][0] = wmma16(av0, b00, accD[0][0]);
    accD[0][1] = wmma16(av0, b01, accD[0][1]);
    accD[1][0] = wmma16(av1, b00, accD[1][0]);
    accD[1][1] = wmma16(av1, b01, accD[1][1]);
  }

  const int half = lane >> 4;
  for (int mi = 0; mi < 2; ++mi)
    for (int ni = 0; ni < 2; ++ni)
      for (int e = 0; e < 8; ++e) {
        const int row = m0 + mi * 16 + half * 8 + e;  // visible index
        const int col = n0 + ni * 16 + r;             // hidden index
        wgrad[(size_t)row * Hn + col] = accM[mi][ni][e] - accD[mi][ni][e];
      }
}

// --- bias grads: column sums (model binary f16 minus data f32) --------------
__global__ void k_colsum_diff(const _Float16* __restrict__ model_f16,
                              const float* __restrict__ data_f32,
                              float* __restrict__ out, int rows, int cols) {
  const int j = blockIdx.x * blockDim.x + threadIdx.x;
  float s = 0.0f;
  for (int b = 0; b < rows; ++b) {
    const size_t idx = (size_t)b * cols + j;
    s += (float)model_f16[idx] - data_f32[idx];
  }
  out[j] = s;
}

// ---------------------------------------------------------------------------
extern "C" void kernel_launch(void* const* d_in, const int* in_sizes, int n_in,
                              void* d_out, int out_size, void* d_ws, size_t ws_size,
                              hipStream_t stream) {
  (void)in_sizes; (void)n_in; (void)out_size; (void)ws_size;
  const float* v   = (const float*)d_in[0];   // [B,V]
  const float* w   = (const float*)d_in[1];   // [V,H]
  const float* vb  = (const float*)d_in[2];   // [V]
  const float* hb  = (const float*)d_in[3];   // [H]
  const float* u_h = (const float*)d_in[4];   // [2,B,H]
  const float* u_v = (const float*)d_in[5];   // [1,B,V]

  float* out      = (float*)d_out;
  float* hp_out   = out + OFF_HP;
  float* recon    = out + OFF_RECON;
  float* wgrad    = out + OFF_WG;
  float* vb_grad  = out + OFF_VB;
  float* hb_grad  = out + OFF_HB;

  // workspace carve (f16 elements)
  _Float16* ws = (_Float16*)d_ws;
  size_t off = 0;
  auto take = [&](size_t n) { _Float16* p = ws + off; off += n; return p; };
  const size_t BV = (size_t)Bn * Vn, BH = (size_t)Bn * Hn, VH = (size_t)Vn * Hn;
  _Float16* v_f16  = take(BV);   // v, row-major [B,V]
  _Float16* vT_f16 = take(BV);   // v^T [V,B]
  _Float16* w_f16  = take(VH);   // w [V,H]       (B-operand for h@wT GEMMs)
  _Float16* wT_f16 = take(VH);   // w^T [H,V]     (B-operand for x@w GEMMs)
  _Float16* hp_f16 = take(BH);   // sigmoid probs [B,H]
  _Float16* h0_f16 = take(BH);   // data-phase hidden sample [B,H]
  _Float16* h0T    = take(BH);   // [H,B]
  _Float16* vk_f16 = take(BV);   // model visible sample [B,V]
  _Float16* vkT    = take(BV);   // [V,B]
  _Float16* h1_f16 = take(BH);   // model hidden sample [B,H]
  _Float16* h1T    = take(BH);   // [H,B]

  const dim3 tb(32, 8);

  // 1) f16 conversions + transposes of inputs
  k_cvt_f16<<<4096, 256, 0, stream>>>(v, v_f16, BV);
  k_cvt_f16<<<2048, 256, 0, stream>>>(w, w_f16, VH);
  k_transpose_f16<float><<<dim3(Hn / 32, Vn / 32), tb, 0, stream>>>(w, wT_f16, Vn, Hn);
  k_transpose_f16<float><<<dim3(Vn / 32, Bn / 32), tb, 0, stream>>>(v, vT_f16, Bn, Vn);

  // 2) h_p = sigmoid(hb + v@w); h0 = (h_p > u_h[0])
  k_gemm_h<0><<<dim3(Hn / 64, Bn / 128), 256, 0, stream>>>(
      v_f16, wT_f16, hb, u_h, hp_out, hp_f16, h0_f16, Vn);
  k_transpose_f16<_Float16><<<dim3(Hn / 32, Bn / 32), tb, 0, stream>>>(
      h0_f16, h0T, Bn, Hn);

  // 3) v_pred + recon error (hp path) and vk sample (h0 path), fused
  hipMemsetAsync(recon, 0, sizeof(float), stream);
  k_gemm_vk<<<dim3(Vn / 64, Bn / 128), 256, 0, stream>>>(
      hp_f16, h0_f16, w_f16, vb, v, u_v, vk_f16, recon, Hn);
  k_transpose_f16<_Float16><<<dim3(Vn / 32, Bn / 32), tb, 0, stream>>>(
      vk_f16, vkT, Bn, Vn);

  // 4) h1 = (sigmoid(hb + vk@w) > u_h[1])
  k_gemm_h<1><<<dim3(Hn / 64, Bn / 128), 256, 0, stream>>>(
      vk_f16, wT_f16, hb, u_h + BH, nullptr, nullptr, h1_f16, Vn);
  k_transpose_f16<_Float16><<<dim3(Hn / 32, Bn / 32), tb, 0, stream>>>(
      h1_f16, h1T, Bn, Hn);

  // 5) w_grad = vk^T@h1 - v^T@h0  (outer-product GEMM over K=B)
  k_gemm_wgrad<<<dim3(Hn / 64, Vn / 128), 256, 0, stream>>>(
      vkT, vT_f16, h1T, h0T, wgrad, Bn);

  // 6) bias grads
  k_colsum_diff<<<Vn / 256, 256, 0, stream>>>(vk_f16, v, vb_grad, Bn, Vn);
  k_colsum_diff<<<Hn / 256, 256, 0, stream>>>(h1_f16, hp_out, hb_grad, Bn, Hn);
}
